// RelativeGlobalAttention_9182640079545
// MI455X (gfx1250) — matrix operand order repaired and checked by reading, not compiled, not run on hardware
//
#include <hip/hip_runtime.h>

// ---------------------------------------------------------------------------
// MI455X (gfx1250) relative global attention.
//   - bf16 WMMA (v_wmma_f32_16x16x32_bf16) for QKV GEMMs, QK^T, Q*Er^T, P*V
//   - CDNA5 async global->LDS (global_load_async_to_lds_b128, ASYNCcnt)
//     with double-buffered LDS in the projection GEMM
//   - fragment loads batched ahead of WMMA bursts (hide DS latency)
//   - flash-attention online softmax, wave32, LDS staging
// Sizes: B=2, S=2048, D=1024, H=16, Dh=64
// ---------------------------------------------------------------------------

typedef unsigned short u16;
typedef __attribute__((ext_vector_type(16))) __bf16 v16bf;
typedef __attribute__((ext_vector_type(8)))  float  v8f;

#define S_LEN   2048
#define D_MODEL 1024
#define NHEAD   16
#define DHEAD   64
#define BATCH   2

union FragU { v16bf v; uint4 q[2]; };

static __device__ inline v16bf ldfrag2(const u16* p0, const u16* p1) {
  FragU f;
  f.q[0] = *reinterpret_cast<const uint4*>(p0);
  f.q[1] = *reinterpret_cast<const uint4*>(p1);
  return f.v;
}
static __device__ inline v16bf ldfrag32(const u16* p) {
  FragU f;
  const uint4* q = reinterpret_cast<const uint4*>(p);
  f.q[0] = q[0]; f.q[1] = q[1];
  return f.v;
}

static __device__ inline v8f wmma_bf16(v16bf a, v16bf b, v8f c) {
  // D(16x16,f32) = A(16x32,bf16) * B(32x16,bf16) + C
  return __builtin_amdgcn_wmma_f32_16x16x32_bf16(
      false, a, false, b, (short)0, c, false, false);
}

static __device__ inline u16 f2bf(float x) {
  unsigned int u = __float_as_uint(x);
  unsigned int r = u + 0x7FFFu + ((u >> 16) & 1u);   // round-to-nearest-even
  return (u16)(r >> 16);
}

// Async global->LDS copy of 16 bytes per lane (CDNA5, tracked by ASYNCcnt).
static __device__ inline void async_b128(void* lds_ptr, const u16* gbase,
                                         size_t eoff) {
  unsigned lds_addr = (unsigned)(size_t)lds_ptr;
  unsigned voff = (unsigned)(eoff * 2u);   // bytes
  asm volatile("global_load_async_to_lds_b128 %0, %1, %2"
               :: "v"(lds_addr), "v"(voff), "s"(gbase)
               : "memory");
}
static __device__ inline void wait_async0() {
  asm volatile("s_wait_asynccnt 0x0" ::: "memory");
}

// ------------------------- fp32 -> bf16 convert ----------------------------
__global__ void cvt_f32_bf16(const float* __restrict__ src,
                             u16* __restrict__ dst, int n) {
  for (int i = blockIdx.x * blockDim.x + threadIdx.x; i < n;
       i += gridDim.x * blockDim.x)
    dst[i] = f2bf(src[i]);
}

// ------------------------- fused QKV projection ----------------------------
// out[mat][b,h,s,dh] = bf16( x[b,s,:] . W[mat][e,:] + bias[mat][e] ).
// GEMM M=4096 N=1024 K=1024. WG tile 64x128, K-step 32, 8 waves, async
// double-buffered LDS. Each wave: 1 A-frag reused across 4 WMMAs.
__global__ __launch_bounds__(256)
void qkv_gemm(const u16* __restrict__ xb, const u16* __restrict__ wb,
              const float* __restrict__ bq, const float* __restrict__ bk,
              const float* __restrict__ bv, u16* __restrict__ qkvw) {
  const int mat = blockIdx.z;
  const u16* W = wb + (size_t)mat * (D_MODEL * D_MODEL);
  const float* bias = (mat == 0) ? bq : ((mat == 1) ? bk : bv);
  u16* out = qkvw + (size_t)mat * (BATCH * NHEAD * S_LEN * DHEAD);

  const int m0 = blockIdx.x * 64;
  const int n0 = blockIdx.y * 128;
  const int tid = threadIdx.x;
  const int w = tid >> 5, lane = tid & 31, hi = lane >> 4, lm = lane & 15;
  const int mt = w & 3;        // 16-row sub-tile
  const int ntp = w >> 2;      // 0..1 -> 64-col half

  __shared__ __align__(16) u16 lA[2][64 * 40];
  __shared__ __align__(16) u16 lB[2][128 * 40];

  // staging map: A 64x4 b128-chunks (1/thread), B 128x4 (2/thread)
  const int ra = tid >> 2, ca = (tid & 3) * 8;

  v8f acc[4];
#pragma unroll
  for (int s = 0; s < 4; ++s) acc[s] = (v8f){0, 0, 0, 0, 0, 0, 0, 0};

  // prologue: stage k-tile 0
  async_b128(&lA[0][ra * 40 + ca], xb, (size_t)(m0 + ra) * D_MODEL + ca);
  async_b128(&lB[0][ra * 40 + ca], W,  (size_t)(n0 + ra) * D_MODEL + ca);
  async_b128(&lB[0][(64 + ra) * 40 + ca], W,
             (size_t)(n0 + 64 + ra) * D_MODEL + ca);

  for (int kt = 0; kt < D_MODEL / 32; ++kt) {
    const int cur = kt & 1;
    wait_async0();
    __syncthreads();
    if (kt + 1 < D_MODEL / 32) {       // stream next tile into other buffer
      const int k0 = (kt + 1) * 32, nxt = cur ^ 1;
      async_b128(&lA[nxt][ra * 40 + ca], xb,
                 (size_t)(m0 + ra) * D_MODEL + k0 + ca);
      async_b128(&lB[nxt][ra * 40 + ca], W,
                 (size_t)(n0 + ra) * D_MODEL + k0 + ca);
      async_b128(&lB[nxt][(64 + ra) * 40 + ca], W,
                 (size_t)(n0 + 64 + ra) * D_MODEL + k0 + ca);
    }
    // batch all fragment loads, then burst the WMMAs
    const u16* arow = &lA[cur][(mt * 16 + lm) * 40];
    v16bf af = ldfrag32(arow + hi * 16);
    v16bf bfr[4];
#pragma unroll
    for (int sub = 0; sub < 4; ++sub)
      bfr[sub] = ldfrag32(&lB[cur][(ntp * 64 + sub * 16 + lm) * 40 + hi * 16]);
#pragma unroll
    for (int sub = 0; sub < 4; ++sub)
      acc[sub] = wmma_bf16(af, bfr[sub], acc[sub]);
    __syncthreads();
  }

  // epilogue: add bias, convert, scatter to [b,h,s,dh]
#pragma unroll
  for (int sub = 0; sub < 4; ++sub) {
    const int n = n0 + ntp * 64 + sub * 16 + lm;
    const float bv_ = bias[n];
    const int h = n >> 6, dh = n & 63;
#pragma unroll
    for (int i = 0; i < 8; ++i) {
      const int m = m0 + mt * 16 + hi * 8 + i;
      const int b = m >> 11, s = m & 2047;
      out[(((size_t)(b * NHEAD + h)) * S_LEN + s) * DHEAD + dh] =
          f2bf(acc[sub][i] + bv_);
    }
  }
}

// ----------------------- flash attention w/ Srel ---------------------------
// Grid: (S/64, H, B). Block: 128 thr = 4 waves; wave w owns rows
// [s0+16w, s0+16w+16). Per key-block of 64:
//   G  = Q_tile(16x64) * Er[r0:r0+80]^T   (Srel band, skew read from LDS)
//   Sc = Q_tile * K_blk^T, mask, online softmax, O += P * V_blk.
__global__ __launch_bounds__(128)
void attn_kernel(const u16* __restrict__ Qw, const u16* __restrict__ Kw,
                 const u16* __restrict__ Vw, const u16* __restrict__ Erw,
                 float* __restrict__ out) {
  const int b = blockIdx.z, h = blockIdx.y, qb = blockIdx.x;
  const int s0 = qb * 64;
  const int tid = threadIdx.x;
  const int w = tid >> 5, lane = tid & 31, hi = lane >> 4, lm = lane & 15;
  const int sw = s0 + w * 16;

  const size_t headoff = ((size_t)(b * NHEAD + h)) * S_LEN * DHEAD;
  const u16* Q = Qw + headoff;
  const u16* K = Kw + headoff;
  const u16* V = Vw + headoff;

  __shared__ __align__(16) u16 lK[64 * 72];
  __shared__ __align__(16) u16 lV[64 * 72];        // transposed: [d][j]
  __shared__ float lG[4][16 * 80];                 // Srel band per wave
  __shared__ __align__(16) u16 lP[4][16 * 72];     // P re-pack per wave

  // Q fragments (A operand), loaded once
  v16bf qa[2];
#pragma unroll
  for (int kh = 0; kh < 2; ++kh) {
    const u16* qrow = Q + (size_t)(sw + lm) * DHEAD + kh * 32;
    qa[kh] = ldfrag32(qrow + hi * 16);
  }

  float mrow[8], lrow[8];
  v8f oacc[4];
#pragma unroll
  for (int i = 0; i < 8; ++i) { mrow[i] = -__builtin_inff(); lrow[i] = 0.f; }
#pragma unroll
  for (int t = 0; t < 4; ++t) oacc[t] = (v8f){0, 0, 0, 0, 0, 0, 0, 0};

  const int nblk = qb + 1;
  for (int blk = 0; blk < nblk; ++blk) {
    const int j0 = blk * 64;
    // ---- K tile: async global->LDS (no VGPR roundtrip) ----
#pragma unroll
    for (int it = 0; it < 4; ++it) {
      const int task = tid + it * 128;
      const int r = task >> 3, c8 = (task & 7) * 8;
      async_b128(&lK[r * 72 + c8], K, (size_t)(j0 + r) * DHEAD + c8);
    }
    // ---- V tile: sync load + register transpose into LDS ----
#pragma unroll
    for (int it = 0; it < 4; ++it) {
      const int task = tid + it * 128;
      const int r = task >> 3, c8 = (task & 7) * 8;
      uint4 vq = *reinterpret_cast<const uint4*>(&V[(size_t)(j0 + r) * DHEAD + c8]);
      const u16* vp = reinterpret_cast<const u16*>(&vq);
#pragma unroll
      for (int e = 0; e < 8; ++e) lV[(c8 + e) * 72 + r] = vp[e];
    }
    wait_async0();
    __syncthreads();

    // ---- Srel band: G = Q * Er[r0 .. r0+80)^T  (skew: r = S-1-s+j) ----
    const int r0 = S_LEN - 16 - sw + j0;
    {
      v16bf ef[5][2];
#pragma unroll
      for (int g = 0; g < 5; ++g)
#pragma unroll
        for (int kh = 0; kh < 2; ++kh) {
          int rr = r0 + g * 16 + lm;
          rr = rr < 0 ? 0 : (rr > S_LEN - 1 ? S_LEN - 1 : rr);
          ef[g][kh] = ldfrag32(Erw + (size_t)rr * DHEAD + kh * 32 + hi * 16);
        }
#pragma unroll
      for (int g = 0; g < 5; ++g) {
        v8f gacc = {0, 0, 0, 0, 0, 0, 0, 0};
        gacc = wmma_bf16(qa[0], ef[g][0], gacc);
        gacc = wmma_bf16(qa[1], ef[g][1], gacc);
#pragma unroll
        for (int i = 0; i < 8; ++i)
          lG[w][(hi * 8 + i) * 80 + g * 16 + lm] = gacc[i];
      }
    }
    asm volatile("s_wait_dscnt 0x0" ::: "memory");

    // ---- scores: Sc = Q * K^T  (batch frag loads, then WMMA burst) ----
    v8f sacc[4];
    {
      v16bf kf[4][2];
#pragma unroll
      for (int t = 0; t < 4; ++t)
#pragma unroll
        for (int kh = 0; kh < 2; ++kh)
          kf[t][kh] = ldfrag32(&lK[(t * 16 + lm) * 72 + kh * 32 + hi * 16]);
#pragma unroll
      for (int t = 0; t < 4; ++t) {
        sacc[t] = (v8f){0, 0, 0, 0, 0, 0, 0, 0};
        sacc[t] = wmma_bf16(qa[0], kf[t][0], sacc[t]);
        sacc[t] = wmma_bf16(qa[1], kf[t][1], sacc[t]);
      }
    }

    // ---- add skewed Srel, scale, causal mask, online softmax ----
    float nmax[8];
#pragma unroll
    for (int i = 0; i < 8; ++i) nmax[i] = -__builtin_inff();
#pragma unroll
    for (int t = 0; t < 4; ++t)
#pragma unroll
      for (int i = 0; i < 8; ++i) {
        const int ls = hi * 8 + i;
        const int s = sw + ls;
        const int j = j0 + t * 16 + lm;
        float v = (sacc[t][i] + lG[w][ls * 80 + (15 - ls) + t * 16 + lm]) * 0.125f;
        if (j > s) v = -__builtin_inff();
        sacc[t][i] = v;
        nmax[i] = fmaxf(nmax[i], v);
      }
#pragma unroll
    for (int msk = 1; msk < 16; msk <<= 1)
#pragma unroll
      for (int i = 0; i < 8; ++i)
        nmax[i] = fmaxf(nmax[i], __shfl_xor(nmax[i], msk, 32));

    float alpha[8], rs[8];
#pragma unroll
    for (int i = 0; i < 8; ++i) {
      const float mn = fmaxf(mrow[i], nmax[i]);
      alpha[i] = __expf(mrow[i] - mn);
      mrow[i] = mn;
      rs[i] = 0.f;
    }
#pragma unroll
    for (int t = 0; t < 4; ++t)
#pragma unroll
      for (int i = 0; i < 8; ++i) {
        const float p = __expf(sacc[t][i] - mrow[i]);
        rs[i] += p;
        lP[w][(hi * 8 + i) * 72 + t * 16 + lm] = f2bf(p);
      }
#pragma unroll
    for (int msk = 1; msk < 16; msk <<= 1)
#pragma unroll
      for (int i = 0; i < 8; ++i) rs[i] += __shfl_xor(rs[i], msk, 32);
#pragma unroll
    for (int i = 0; i < 8; ++i) lrow[i] = lrow[i] * alpha[i] + rs[i];
#pragma unroll
    for (int t = 0; t < 4; ++t)
#pragma unroll
      for (int i = 0; i < 8; ++i) oacc[t][i] *= alpha[i];
    asm volatile("s_wait_dscnt 0x0" ::: "memory");

    // ---- O += P * V  (batch frag loads, then WMMA burst) ----
    {
      v16bf pa[2], vf[4][2];
#pragma unroll
      for (int kh = 0; kh < 2; ++kh) {
        const u16* pr = &lP[w][lm * 72 + kh * 32];
        pa[kh] = ldfrag32(pr + hi * 16);
      }
#pragma unroll
      for (int t = 0; t < 4; ++t)
#pragma unroll
        for (int kh = 0; kh < 2; ++kh)
          vf[t][kh] = ldfrag32(&lV[(t * 16 + lm) * 72 + kh * 32 + hi * 16]);
#pragma unroll
      for (int t = 0; t < 4; ++t) {
        oacc[t] = wmma_bf16(pa[0], vf[t][0], oacc[t]);
        oacc[t] = wmma_bf16(pa[1], vf[t][1], oacc[t]);
      }
    }
    __syncthreads();
  }

  // ---- epilogue: O / l, scatter to [b, s, h*64+d] fp32 ----
#pragma unroll
  for (int t = 0; t < 4; ++t)
#pragma unroll
    for (int i = 0; i < 8; ++i) {
      const int ls = hi * 8 + i;
      const int s = sw + ls;
      out[((size_t)b * S_LEN + s) * D_MODEL + h * DHEAD + t * 16 + lm] =
          oacc[t][i] / lrow[i];
    }
}

// ---------------------------------------------------------------------------
extern "C" void kernel_launch(void* const* d_in, const int* in_sizes, int n_in,
                              void* d_out, int out_size, void* d_ws, size_t ws_size,
                              hipStream_t stream) {
  (void)in_sizes; (void)n_in; (void)out_size; (void)ws_size;
  const float* x  = (const float*)d_in[0];
  const float* Wq = (const float*)d_in[1];
  const float* bq = (const float*)d_in[2];
  const float* Wk = (const float*)d_in[3];
  const float* bk = (const float*)d_in[4];
  const float* Wv = (const float*)d_in[5];
  const float* bv = (const float*)d_in[6];
  const float* Er = (const float*)d_in[7];

  char* ws = (char*)d_ws;
  u16* xb   = (u16*)(ws);                          // 8 MB
  u16* wb   = (u16*)(ws + (size_t)8  * 1048576);   // 3 x 2 MB
  u16* erb  = (u16*)(ws + (size_t)14 * 1048576);   // 256 KB
  u16* qkvw = (u16*)(ws + (size_t)16 * 1048576);   // 3 x 8 MB (Q, K, V)

  const int NX = BATCH * S_LEN * D_MODEL;          // 4194304
  const int NW = D_MODEL * D_MODEL;                // 1048576
  const int NE = S_LEN * DHEAD;                    // 131072

  cvt_f32_bf16<<<2048, 256, 0, stream>>>(x,  xb, NX);
  cvt_f32_bf16<<<1024, 256, 0, stream>>>(Wq, wb + 0 * (size_t)NW, NW);
  cvt_f32_bf16<<<1024, 256, 0, stream>>>(Wk, wb + 1 * (size_t)NW, NW);
  cvt_f32_bf16<<<1024, 256, 0, stream>>>(Wv, wb + 2 * (size_t)NW, NW);
  cvt_f32_bf16<<<256,  256, 0, stream>>>(Er, erb, NE);

  qkv_gemm<<<dim3(64, 8, 3), 256, 0, stream>>>(xb, wb, bq, bk, bv, qkvw);

  const size_t HSZ = (size_t)BATCH * NHEAD * S_LEN * DHEAD;  // 4194304 elems
  attn_kernel<<<dim3(S_LEN / 64, NHEAD, BATCH), 128, 0, stream>>>(
      qkvw, qkvw + HSZ, qkvw + 2 * HSZ, erb, (float*)d_out);
}
